// ComplexAttention_65094524339042
// MI455X (gfx1250) — matrix operand order repaired and checked by reading, not compiled, not run on hardware
//
#include <hip/hip_runtime.h>
#include <hip/hip_bf16.h>

// Problem constants (from reference): B=8, C=512, H=W=32 -> T=1024, dh=64.
#define KB 8
#define KC 512
#define KT 1024
#define SCALE 0.125f  // 1/sqrt(dh) = 1/8

typedef __attribute__((ext_vector_type(16))) __bf16 v16bf;
typedef __attribute__((ext_vector_type(8)))  float  v8f;

static __device__ __forceinline__ v8f wmma_bf16(v16bf a, v16bf b, v8f c) {
  // D = A(16x32 bf16) * B(32x16 bf16) + C(16x16 f32)
  return __builtin_amdgcn_wmma_f32_16x16x32_bf16(false, a, false, b, (short)0, c,
                                                 false, false);
}

// A-fragment K index for element e given lane-half (contiguous: k = 16*half + e, matches B)
static __device__ __forceinline__ int a_k(int e, int half) {
  return (half << 4) + e;
}

// ---------------------------------------------------------------------------
// 1) Projection + rotation:  out_{re,im}[b,t,d] = rot( Z @ W^T , phi )
//    Z is [B, C, T] (channel-major, as given); outputs bf16 [B, T, C].
//    One wave per 32(t) x 32(d) tile, 2x2 register blocking (8 WMMA / K-step).
// ---------------------------------------------------------------------------
__global__ __launch_bounds__(32)
void proj_kernel(const float* __restrict__ zre, const float* __restrict__ zim,
                 const float* __restrict__ Wm,  const float* __restrict__ phi,
                 __bf16* __restrict__ out_re, __bf16* __restrict__ out_im) {
  const int t0 = blockIdx.x << 5;
  const int d0 = blockIdx.y << 5;
  const int b  = blockIdx.z;
  const int l    = threadIdx.x & 31;
  const int half = l >> 4;
  const int lm   = l & 15;

  const float* zre_b = zre + (size_t)b * KC * KT;
  const float* zim_b = zim + (size_t)b * KC * KT;

  v8f acc_re[2][2] = {{{}, {}}, {{}, {}}};
  v8f acc_im[2][2] = {{{}, {}}, {{}, {}}};

  for (int c0 = 0; c0 < KC; c0 += 32) {
    // A[i][m,k] = Z[b, c0+k, t0+16i+m]; per-e instruction is lane-coalesced
    v16bf a_re[2], a_im[2];
#pragma unroll
    for (int i = 0; i < 2; ++i) {
#pragma unroll
      for (int e = 0; e < 16; ++e) {
        int k = a_k(e, half);
        size_t idx = (size_t)(c0 + k) * KT + (t0 + (i << 4) + lm);
        a_re[i][e] = (__bf16)zre_b[idx];
        a_im[i][e] = (__bf16)zim_b[idx];
      }
    }
    // B[j][k,n] = W[d0+16j+n, c0+k] -> 16 contiguous floats per lane
    v16bf bw[2];
#pragma unroll
    for (int j = 0; j < 2; ++j) {
      const float* wrow = Wm + (size_t)(d0 + (j << 4) + lm) * KC + c0 + (half << 4);
#pragma unroll
      for (int e = 0; e < 16; ++e) bw[j][e] = (__bf16)wrow[e];
    }
#pragma unroll
    for (int i = 0; i < 2; ++i)
#pragma unroll
      for (int j = 0; j < 2; ++j) {
        acc_re[i][j] = wmma_bf16(a_re[i], bw[j], acc_re[i][j]);
        acc_im[i][j] = wmma_bf16(a_im[i], bw[j], acc_im[i][j]);
      }
  }

  __bf16* ore = out_re + (size_t)b * KT * KC;
  __bf16* oim = out_im + (size_t)b * KT * KC;
#pragma unroll
  for (int j = 0; j < 2; ++j) {
    const int d = d0 + (j << 4) + lm;       // fixed output column per lane
    const float ph = phi[d];
    const float cp = __cosf(ph), sp = __sinf(ph);
#pragma unroll
    for (int i = 0; i < 2; ++i)
#pragma unroll
      for (int v = 0; v < 8; ++v) {
        int t = t0 + (i << 4) + (half << 3) + v;
        float re2 = acc_re[i][j][v], im2 = acc_im[i][j][v];
        ore[(size_t)t * KC + d] = (__bf16)(re2 * cp - im2 * sp);
        oim[(size_t)t * KC + d] = (__bf16)(re2 * sp + im2 * cp);
      }
  }
}

// ---------------------------------------------------------------------------
// 2) Scores: S[b,t,u] = (Qre.Kre + Qim.Kim)/8 with causal mask (u>t -> -1e30)
//    Q,K bf16 [B,T,C]; S fp32 [B,T,T]. One wave per 32x32 S tile (2x2 block).
// ---------------------------------------------------------------------------
__global__ __launch_bounds__(32)
void scores_kernel(const __bf16* __restrict__ qre, const __bf16* __restrict__ qim,
                   const __bf16* __restrict__ kre, const __bf16* __restrict__ kim,
                   float* __restrict__ S) {
  const int t0 = blockIdx.x << 5;
  const int u0 = blockIdx.y << 5;
  const int b  = blockIdx.z;
  const int l = threadIdx.x & 31, half = l >> 4, lm = l & 15;
  float* Sb = S + (size_t)b * KT * KT;

  if (u0 > t0) {  // super-tile entirely above the diagonal: masked, no GEMM
#pragma unroll
    for (int i = 0; i < 2; ++i)
#pragma unroll
      for (int v = 0; v < 8; ++v) {
        int t = t0 + (i << 4) + (half << 3) + v;
#pragma unroll
        for (int j = 0; j < 2; ++j)
          Sb[(size_t)t * KT + u0 + (j << 4) + lm] = -1e30f;
      }
    return;
  }

  const __bf16* qrb = qre + (size_t)b * KT * KC;
  const __bf16* qib = qim + (size_t)b * KT * KC;
  const __bf16* krb = kre + (size_t)b * KT * KC;
  const __bf16* kib = kim + (size_t)b * KT * KC;

  v8f acc[2][2] = {{{}, {}}, {{}, {}}};
  for (int c0 = 0; c0 < KC; c0 += 32) {
    // A[i][m,k] = Q[t0+16i+m, c0+k] (row-major: one 16-elem run per lane)
    v16bf aqr[2], aqi[2];
#pragma unroll
    for (int i = 0; i < 2; ++i) {
#pragma unroll
      for (int e = 0; e < 16; ++e) {
        int k = a_k(e, half);
        size_t idx = (size_t)(t0 + (i << 4) + lm) * KC + c0 + k;
        aqr[i][e] = qrb[idx];
        aqi[i][e] = qib[idx];
      }
    }
    // B[j][k,n] = K[u0+16j+n, c0+k] -> 16 contiguous bf16 per lane
    v16bf bkr[2], bki[2];
#pragma unroll
    for (int j = 0; j < 2; ++j) {
      size_t kbase = (size_t)(u0 + (j << 4) + lm) * KC + c0 + (half << 4);
#pragma unroll
      for (int e = 0; e < 16; ++e) { bkr[j][e] = krb[kbase + e]; bki[j][e] = kib[kbase + e]; }
    }
#pragma unroll
    for (int i = 0; i < 2; ++i)
#pragma unroll
      for (int j = 0; j < 2; ++j) {
        acc[i][j] = wmma_bf16(aqr[i], bkr[j], acc[i][j]);  // real part
        acc[i][j] = wmma_bf16(aqi[i], bki[j], acc[i][j]);  // + imaginary part
      }
  }

#pragma unroll
  for (int i = 0; i < 2; ++i)
#pragma unroll
    for (int v = 0; v < 8; ++v) {
      int t = t0 + (i << 4) + (half << 3) + v;
#pragma unroll
      for (int j = 0; j < 2; ++j) {
        int u = u0 + (j << 4) + lm;
        float val = acc[i][j][v] * SCALE;
        if (u > t) val = -1e30f;
        Sb[(size_t)t * KT + u] = val;
      }
    }
}

// ---------------------------------------------------------------------------
// 3) Row softmax in place over S [B*T rows, T cols]. One 256-thread block/row.
// ---------------------------------------------------------------------------
__global__ __launch_bounds__(256)
void softmax_kernel(float* __restrict__ S) {
  __shared__ float red[256];
  const int tid = threadIdx.x;
  float* r = S + (size_t)blockIdx.x * KT;

  float vals[4];
  float m = -1e30f;
#pragma unroll
  for (int i = 0; i < 4; ++i) { vals[i] = r[tid + (i << 8)]; m = fmaxf(m, vals[i]); }
  red[tid] = m; __syncthreads();
  for (int s = 128; s > 0; s >>= 1) {
    if (tid < s) red[tid] = fmaxf(red[tid], red[tid + s]);
    __syncthreads();
  }
  m = red[0]; __syncthreads();

  float sum = 0.f;
#pragma unroll
  for (int i = 0; i < 4; ++i) { vals[i] = __expf(vals[i] - m); sum += vals[i]; }
  red[tid] = sum; __syncthreads();
  for (int s = 128; s > 0; s >>= 1) {
    if (tid < s) red[tid] += red[tid + s];
    __syncthreads();
  }
  const float inv = 1.0f / red[0];
#pragma unroll
  for (int i = 0; i < 4; ++i) r[tid + (i << 8)] = vals[i] * inv;
}

// ---------------------------------------------------------------------------
// 4) O = P @ V (re and im share the P A-fragments). Causal: stop the K loop at
//    the query tile's diagonal (masked P entries are exactly 0 after softmax).
//    P fp32 [B,T,T]; V bf16 [B,T,C]; O bf16 [B,T,C]. 32x32 tile per wave.
// ---------------------------------------------------------------------------
__global__ __launch_bounds__(32)
void pv_kernel(const float* __restrict__ P,
               const __bf16* __restrict__ vre, const __bf16* __restrict__ vim,
               __bf16* __restrict__ ore, __bf16* __restrict__ oim) {
  const int t0 = blockIdx.x << 5;
  const int c0 = blockIdx.y << 5;
  const int b  = blockIdx.z;
  const int l = threadIdx.x & 31, half = l >> 4, lm = l & 15;

  const float*  Pb = P   + (size_t)b * KT * KT;
  const __bf16* vr = vre + (size_t)b * KT * KC;
  const __bf16* vi = vim + (size_t)b * KT * KC;

  v8f acc_r[2][2] = {{{}, {}}, {{}, {}}};
  v8f acc_i[2][2] = {{{}, {}}, {{}, {}}};

  for (int u0 = 0; u0 < t0 + 32; u0 += 32) {
    // A[i][m,k] = P[t0+16i+m, u0+k]
    v16bf ap[2];
#pragma unroll
    for (int i = 0; i < 2; ++i) {
#pragma unroll
      for (int e = 0; e < 16; ++e) {
        int k = a_k(e, half);
        ap[i][e] = (__bf16)Pb[(size_t)(t0 + (i << 4) + lm) * KT + u0 + k];
      }
    }
    // B[j][k,n] = V[u0+k, c0+16j+n]  (k-strided by C)
    v16bf bvr[2], bvi[2];
#pragma unroll
    for (int j = 0; j < 2; ++j) {
#pragma unroll
      for (int e = 0; e < 16; ++e) {
        size_t idx = (size_t)(u0 + (half << 4) + e) * KC + c0 + (j << 4) + lm;
        bvr[j][e] = vr[idx];
        bvi[j][e] = vi[idx];
      }
    }
#pragma unroll
    for (int i = 0; i < 2; ++i)
#pragma unroll
      for (int j = 0; j < 2; ++j) {
        acc_r[i][j] = wmma_bf16(ap[i], bvr[j], acc_r[i][j]);
        acc_i[i][j] = wmma_bf16(ap[i], bvi[j], acc_i[i][j]);
      }
  }

  __bf16* po = ore + (size_t)b * KT * KC;
  __bf16* qo = oim + (size_t)b * KT * KC;
#pragma unroll
  for (int i = 0; i < 2; ++i)
#pragma unroll
    for (int v = 0; v < 8; ++v) {
      int t = t0 + (i << 4) + (half << 3) + v;
#pragma unroll
      for (int j = 0; j < 2; ++j) {
        int c = c0 + (j << 4) + lm;
        po[(size_t)t * KC + c] = (__bf16)acc_r[i][j][v];
        qo[(size_t)t * KC + c] = (__bf16)acc_i[i][j][v];
      }
    }
}

// ---------------------------------------------------------------------------
// 5) Output projection + rotation, written as fp32 [B, C, H, W] (= [B,C,T]):
//    y[b,d,t] = rot( sum_c O[b,t,c] * Wo[d,c] , phi_o[d] )
//    M = d (channels), N = t (tokens); 32x32 tile per wave.
// ---------------------------------------------------------------------------
__global__ __launch_bounds__(32)
void outproj_kernel(const __bf16* __restrict__ ore, const __bf16* __restrict__ oim,
                    const float* __restrict__ Wo, const float* __restrict__ phi,
                    float* __restrict__ yre, float* __restrict__ yim) {
  const int d0 = blockIdx.x << 5;
  const int t0 = blockIdx.y << 5;
  const int b  = blockIdx.z;
  const int l = threadIdx.x & 31, half = l >> 4, lm = l & 15;

  const __bf16* orb = ore + (size_t)b * KT * KC;
  const __bf16* oib = oim + (size_t)b * KT * KC;

  v8f acc_r[2][2] = {{{}, {}}, {{}, {}}};
  v8f acc_i[2][2] = {{{}, {}}, {{}, {}}};

  for (int c0 = 0; c0 < KC; c0 += 32) {
    // A[i][m,k] = Wo[d0+16i+m, c0+k] (row-major fp32)
    v16bf aw[2];
#pragma unroll
    for (int i = 0; i < 2; ++i) {
#pragma unroll
      for (int e = 0; e < 16; ++e) {
        int k = a_k(e, half);
        aw[i][e] = (__bf16)Wo[(size_t)(d0 + (i << 4) + lm) * KC + c0 + k];
      }
    }
    // B[j][k,n] = O[b, t0+16j+n, c0+k] -> contiguous bf16 per lane
    v16bf br[2], bi[2];
#pragma unroll
    for (int j = 0; j < 2; ++j) {
      size_t obase = (size_t)(t0 + (j << 4) + lm) * KC + c0 + (half << 4);
#pragma unroll
      for (int e = 0; e < 16; ++e) { br[j][e] = orb[obase + e]; bi[j][e] = oib[obase + e]; }
    }
#pragma unroll
    for (int i = 0; i < 2; ++i)
#pragma unroll
      for (int j = 0; j < 2; ++j) {
        acc_r[i][j] = wmma_bf16(aw[i], br[j], acc_r[i][j]);
        acc_i[i][j] = wmma_bf16(aw[i], bi[j], acc_i[i][j]);
      }
  }

  float* yr = yre + (size_t)b * KC * KT;
  float* yi = yim + (size_t)b * KC * KT;
#pragma unroll
  for (int i = 0; i < 2; ++i)
#pragma unroll
    for (int v = 0; v < 8; ++v) {
      int d = d0 + (i << 4) + (half << 3) + v;  // output channel per element
      float ph = phi[d];
      float cp = __cosf(ph), sp = __sinf(ph);
#pragma unroll
      for (int j = 0; j < 2; ++j) {
        int t = t0 + (j << 4) + lm;
        float re2 = acc_r[i][j][v], im2 = acc_i[i][j][v];
        yr[(size_t)d * KT + t] = re2 * cp - im2 * sp;
        yi[(size_t)d * KT + t] = re2 * sp + im2 * cp;
      }
    }
}

// ---------------------------------------------------------------------------
extern "C" void kernel_launch(void* const* d_in, const int* in_sizes, int n_in,
                              void* d_out, int out_size, void* d_ws, size_t ws_size,
                              hipStream_t stream) {
  const float* zre  = (const float*)d_in[0];
  const float* zim  = (const float*)d_in[1];
  const float* Wq   = (const float*)d_in[2];
  const float* phiq = (const float*)d_in[3];
  const float* Wk   = (const float*)d_in[4];
  const float* phik = (const float*)d_in[5];
  const float* Wv   = (const float*)d_in[6];
  const float* phiv = (const float*)d_in[7];
  const float* Wo   = (const float*)d_in[8];
  const float* phio = (const float*)d_in[9];

  const size_t ntc = (size_t)KB * KT * KC;          // 4M elements
  char* ws = (char*)d_ws;
  __bf16* qre = (__bf16*)ws; ws += ntc * sizeof(__bf16);
  __bf16* qim = (__bf16*)ws; ws += ntc * sizeof(__bf16);
  __bf16* kre = (__bf16*)ws; ws += ntc * sizeof(__bf16);
  __bf16* kim = (__bf16*)ws; ws += ntc * sizeof(__bf16);
  __bf16* vre = (__bf16*)ws; ws += ntc * sizeof(__bf16);
  __bf16* vim = (__bf16*)ws; ws += ntc * sizeof(__bf16);
  __bf16* ore = (__bf16*)ws; ws += ntc * sizeof(__bf16);
  __bf16* oim = (__bf16*)ws; ws += ntc * sizeof(__bf16);
  float*  S   = (float*)ws;                          // [B, T, T] fp32 (32 MB)

  float* out_re = (float*)d_out;
  float* out_im = out_re + ntc;

  const dim3 wave(32);
  const dim3 gProj(KT / 32, KC / 32, KB);
  proj_kernel<<<gProj, wave, 0, stream>>>(zre, zim, Wq, phiq, qre, qim);
  proj_kernel<<<gProj, wave, 0, stream>>>(zre, zim, Wk, phik, kre, kim);
  proj_kernel<<<gProj, wave, 0, stream>>>(zre, zim, Wv, phiv, vre, vim);

  scores_kernel<<<dim3(KT / 32, KT / 32, KB), wave, 0, stream>>>(qre, qim, kre, kim, S);
  softmax_kernel<<<dim3(KB * KT), dim3(256), 0, stream>>>(S);
  pv_kernel<<<dim3(KT / 32, KC / 32, KB), wave, 0, stream>>>(S, vre, vim, ore, oim);
  outproj_kernel<<<dim3(KC / 32, KT / 32, KB), wave, 0, stream>>>(ore, oim, Wo, phio,
                                                                 out_re, out_im);
}
